// MultiHeadAttention_412316860549
// MI455X (gfx1250) — matrix operand order repaired and checked by reading, not compiled, not run on hardware
//
#include <hip/hip_runtime.h>
#include <hip/hip_bf16.h>

// ---------- types ----------
typedef __attribute__((ext_vector_type(16))) __bf16 v16bf;
typedef __attribute__((ext_vector_type(8)))  __bf16 v8bf;
typedef __attribute__((ext_vector_type(8)))  float  v8f;

union V16U { v16bf v; v8bf h[2]; };

__device__ __forceinline__ __bf16 f32_to_bf16(float f) {
  union { float f; unsigned u; } c; c.f = f;
  unsigned u = c.u;
  u += 0x7FFFu + ((u >> 16) & 1u);               // round-to-nearest-even
  unsigned short s = (unsigned short)(u >> 16);
  union { unsigned short s; __bf16 b; } r; r.s = s;
  return r.b;
}

// ---------- WMMA fragment gathers (wave32, 16x16x32 bf16) ----------
// A (16xK, row-major, rows m=row0+lo):
//   lanes 0-15 : K = 0..7 (vgpr0-3), 16..23 (vgpr4-7)
//   lanes 16-31: K = 8..15,          24..31
__device__ __forceinline__ v16bf frag_a(const __bf16* base, int stride, int row0, int lane) {
  int lo = lane & 15, hi = lane >> 4;
  const __bf16* p = base + (size_t)(row0 + lo) * stride + 8 * hi;
  V16U r;
  r.h[0] = *reinterpret_cast<const v8bf*>(p);
  r.h[1] = *reinterpret_cast<const v8bf*>(p + 16);
  return r.v;
}
// B (Kx16) stored N-major in LDS (row n contiguous in K), cols n=row0+lo:
//   lanes 0-15 hold K=0..7,16..23, lanes 16-31 hold K=8..15,24..31
__device__ __forceinline__ v16bf frag_b(const __bf16* base, int stride, int row0, int lane) {
  int lo = lane & 15, hi = lane >> 4;
  const __bf16* p = base + (size_t)(row0 + lo) * stride + 8 * hi;
  V16U r;
  r.h[0] = *reinterpret_cast<const v8bf*>(p);
  r.h[1] = *reinterpret_cast<const v8bf*>(p + 16);
  return r.v;
}

// ---------- fp32 -> bf16 conversion (4 elems/thread) ----------
struct alignas(8) BF4 { __bf16 a, b, c, d; };
__global__ void cvt_f32_bf16(const float* __restrict__ x, __bf16* __restrict__ y, int n4) {
  int i = blockIdx.x * blockDim.x + threadIdx.x;
  if (i >= n4) return;
  float4 f = reinterpret_cast<const float4*>(x)[i];
  BF4 o; o.a = f32_to_bf16(f.x); o.b = f32_to_bf16(f.y);
  o.c = f32_to_bf16(f.z); o.d = f32_to_bf16(f.w);
  reinterpret_cast<BF4*>(y)[i] = o;
}

// ---------- weight transpose + convert: Wt[n][k] = bf16(W[k][n]) ----------
__global__ void transpose_w(const float* __restrict__ W, __bf16* __restrict__ Wt, int Dd) {
  int n = blockIdx.x * 16 + threadIdx.x;
  int k = blockIdx.y * 16 + threadIdx.y;
  Wt[(size_t)n * Dd + k] = f32_to_bf16(W[(size_t)k * Dd + n]);
}

// ---------- bf16 GEMM: C[M,N] = alpha * A[M,K] @ Bt[N,K]^T ----------
// block tile 128x128x32, 256 threads = 8 waves (2 M x 4 N), wave tile 64x32
template <bool BF16OUT>
__global__ __launch_bounds__(256) void gemm_bf16(
    const __bf16* __restrict__ A, const __bf16* __restrict__ Bt,
    void* __restrict__ Cout, int M, int N, int K, float alpha) {
  constexpr int BM = 128, BN = 128, BK = 32;
  __shared__ __align__(32) __bf16 sA[BM * BK];
  __shared__ __align__(32) __bf16 sB[BN * BK];

  const int tid  = threadIdx.x;
  const int lane = tid & 31, wave = tid >> 5;
  const int wm = wave >> 2, wn = wave & 3;
  const int tileM = blockIdx.y * BM, tileN = blockIdx.x * BN;

  v8f acc[4][2];
  for (int i = 0; i < 4; ++i)
    for (int j = 0; j < 2; ++j)
      for (int e = 0; e < 8; ++e) acc[i][j][e] = 0.f;

  const int lrow = tid >> 2;          // 0..63
  const int lcol = (tid & 3) * 8;     // 0,8,16,24

  for (int kk = 0; kk < K; kk += BK) {
    if (kk + BK < K) {                // prefetch next K-tile -> global_prefetch_b8
      __builtin_prefetch(A + (size_t)(tileM + lrow) * K + kk + BK + lcol, 0, 1);
      __builtin_prefetch(Bt + (size_t)(tileN + lrow) * K + kk + BK + lcol, 0, 1);
    }
    for (int p = 0; p < 2; ++p) {
      int r = p * 64 + lrow;
      *reinterpret_cast<v8bf*>(sA + r * BK + lcol) =
          *reinterpret_cast<const v8bf*>(A + (size_t)(tileM + r) * K + kk + lcol);
      *reinterpret_cast<v8bf*>(sB + r * BK + lcol) =
          *reinterpret_cast<const v8bf*>(Bt + (size_t)(tileN + r) * K + kk + lcol);
    }
    __syncthreads();

    v16bf af[4], bf[2];
    for (int i = 0; i < 4; ++i) af[i] = frag_a(sA, BK, wm * 64 + i * 16, lane);
    for (int j = 0; j < 2; ++j) bf[j] = frag_b(sB, BK, wn * 32 + j * 16, lane);
    for (int i = 0; i < 4; ++i)
      for (int j = 0; j < 2; ++j)
        acc[i][j] = __builtin_amdgcn_wmma_f32_16x16x32_bf16(
            false, af[i], false, bf[j], (short)0, acc[i][j], false, false);
    __syncthreads();
  }

  const int lo = lane & 15, hi = lane >> 4;
  for (int i = 0; i < 4; ++i)
    for (int j = 0; j < 2; ++j) {
      int col  = tileN + wn * 32 + j * 16 + lo;
      int rowb = tileM + wm * 64 + i * 16 + 8 * hi;   // + r below
      for (int r = 0; r < 8; ++r) {
        float v = acc[i][j][r] * alpha;
        size_t idx = (size_t)(rowb + r) * N + col;
        if constexpr (BF16OUT) reinterpret_cast<__bf16*>(Cout)[idx] = f32_to_bf16(v);
        else                   reinterpret_cast<float*>(Cout)[idx]  = v;
      }
    }
}

// ---------- flash attention over heads: 1 block = 64 query rows of one (b,h) ----------
// 128 threads = 4 waves, each wave owns 16 query rows; stream 64-wide K/V tiles.
__global__ __launch_bounds__(128) void flash_attn(
    const __bf16* __restrict__ Q, const __bf16* __restrict__ Kx,
    const __bf16* __restrict__ V, __bf16* __restrict__ O) {
  constexpr int S = 2048, D = 1024, DK = 64;
  const int b = blockIdx.y >> 4;       // H = 16
  const int h = blockIdx.y & 15;
  const int tid = threadIdx.x, lane = tid & 31, wave = tid >> 5;
  const int lo = lane & 15, hi = lane >> 4;
  const int q0 = blockIdx.x * 64 + wave * 16;

  __shared__ __align__(32) __bf16 kt[64 * 64];        // K tile [s][d]
  __shared__ __align__(32) __bf16 vt[64 * 64];        // V tile transposed [d][s]
  __shared__ __align__(32) __bf16 pb[4][16 * 64];     // per-wave P staging

  const size_t rowbase = (size_t)b * S;
  const __bf16* Qb = Q + rowbase * D + h * DK;

  // Q A-fragments (16 rows x 64 d = 2 K-steps), loaded straight from global
  v16bf qf[2];
  for (int ks = 0; ks < 2; ++ks) {
    const __bf16* p = Qb + (size_t)(q0 + lo) * D + ks * 32 + 8 * hi;
    V16U r;
    r.h[0] = *reinterpret_cast<const v8bf*>(p);
    r.h[1] = *reinterpret_cast<const v8bf*>(p + 16);
    qf[ks] = r.v;
  }

  float mst[8], lst[8];
  v8f ov[4];
  for (int r = 0; r < 8; ++r) { mst[r] = -1e30f; lst[r] = 0.f; }
  for (int j = 0; j < 4; ++j)
    for (int e = 0; e < 8; ++e) ov[j][e] = 0.f;

  const int lrow = tid >> 3;           // 0..15
  const int lcol = (tid & 7) * 8;      // 0..56

  for (int kt0 = 0; kt0 < S; kt0 += 64) {
    const __bf16* Kb = Kx + (rowbase + kt0) * D + h * DK;
    const __bf16* Vb = V  + (rowbase + kt0) * D + h * DK;
    for (int p = 0; p < 4; ++p) {
      int s = p * 16 + lrow;
      *reinterpret_cast<v8bf*>(kt + s * 64 + lcol) =
          *reinterpret_cast<const v8bf*>(Kb + (size_t)s * D + lcol);
      v8bf vv = *reinterpret_cast<const v8bf*>(Vb + (size_t)s * D + lcol);
      for (int e = 0; e < 8; ++e) vt[(lcol + e) * 64 + s] = vv[e];  // transpose
    }
    __syncthreads();

    // scores S = Q K^T : 16x64 per wave, 4 N-subtiles x 2 K-steps
    v8f sc[4];
    for (int nj = 0; nj < 4; ++nj) {
      v8f c; for (int e = 0; e < 8; ++e) c[e] = 0.f;
      for (int ks = 0; ks < 2; ++ks) {
        v16bf kf = frag_b(kt + ks * 32, 64, nj * 16, lane);
        c = __builtin_amdgcn_wmma_f32_16x16x32_bf16(
            false, qf[ks], false, kf, (short)0, c, false, false);
      }
      sc[nj] = c;
    }

    // online softmax; C-layout: element r <-> row (r + 8*hi), col (nj*16 + lo)
    float rmax[8];
    for (int r = 0; r < 8; ++r)
      rmax[r] = fmaxf(fmaxf(sc[0][r], sc[1][r]), fmaxf(sc[2][r], sc[3][r]));
    for (int m = 1; m < 16; m <<= 1)
      for (int r = 0; r < 8; ++r)
        rmax[r] = fmaxf(rmax[r], __shfl_xor(rmax[r], m, 32));

    float corr[8], rsum[8];
    for (int r = 0; r < 8; ++r) {
      float mn = fmaxf(mst[r], rmax[r]);
      corr[r] = __expf(mst[r] - mn);
      mst[r] = mn;
      rsum[r] = 0.f;
    }
    for (int nj = 0; nj < 4; ++nj)
      for (int r = 0; r < 8; ++r) {
        float p = __expf(sc[nj][r] - mst[r]);
        sc[nj][r] = p;
        rsum[r] += p;
      }
    for (int m = 1; m < 16; m <<= 1)
      for (int r = 0; r < 8; ++r)
        rsum[r] += __shfl_xor(rsum[r], m, 32);
    for (int r = 0; r < 8; ++r) lst[r] = lst[r] * corr[r] + rsum[r];
    for (int j = 0; j < 4; ++j)
      for (int r = 0; r < 8; ++r) ov[j][r] *= corr[r];

    // P: C-layout -> A-layout via per-wave LDS round-trip
    __bf16* pw = pb[wave];
    for (int nj = 0; nj < 4; ++nj)
      for (int r = 0; r < 8; ++r)
        pw[(r + 8 * hi) * 64 + nj * 16 + lo] = f32_to_bf16(sc[nj][r]);
    asm volatile("s_wait_dscnt 0" ::: "memory");     // same-wave LDS RAW

    v16bf pf[2];
    for (int ks = 0; ks < 2; ++ks) pf[ks] = frag_a(pw + ks * 32, 64, 0, lane);
    for (int dj = 0; dj < 4; ++dj)
      for (int ks = 0; ks < 2; ++ks) {
        v16bf vf = frag_b(vt + ks * 32, 64, dj * 16, lane);
        ov[dj] = __builtin_amdgcn_wmma_f32_16x16x32_bf16(
            false, pf[ks], false, vf, (short)0, ov[dj], false, false);
      }
    __syncthreads();
  }

  // O = O / l, write bf16 into per-head slice of attn_out
  __bf16* Ob = O + (rowbase + q0) * D + h * DK;
  for (int r = 0; r < 8; ++r) {
    float inv = 1.0f / lst[r];
    for (int dj = 0; dj < 4; ++dj)
      Ob[(size_t)(r + 8 * hi) * D + dj * 16 + lo] = f32_to_bf16(ov[dj][r] * inv);
  }
}

// ---------- launch ----------
extern "C" void kernel_launch(void* const* d_in, const int* in_sizes, int n_in,
                              void* d_out, int out_size, void* d_ws, size_t ws_size,
                              hipStream_t stream) {
  (void)in_sizes; (void)n_in; (void)out_size; (void)ws_size;
  const float* q  = (const float*)d_in[0];
  const float* k  = (const float*)d_in[1];
  const float* v  = (const float*)d_in[2];
  const float* Wq = (const float*)d_in[3];
  const float* Wk = (const float*)d_in[4];
  const float* Wv = (const float*)d_in[5];
  const float* Wo = (const float*)d_in[6];

  constexpr int Bc = 2, Sc = 2048, Dc = 1024;
  constexpr size_t Mrows = (size_t)Bc * Sc;   // 4096
  constexpr size_t MSD   = Mrows * Dc;        // 4,194,304
  constexpr size_t DD    = (size_t)Dc * Dc;

  char* w = (char*)d_ws;
  __bf16* xq  = (__bf16*)w; w += MSD * 2;
  __bf16* xk  = (__bf16*)w; w += MSD * 2;
  __bf16* xv  = (__bf16*)w; w += MSD * 2;
  __bf16* wqt = (__bf16*)w; w += DD * 2;
  __bf16* wkt = (__bf16*)w; w += DD * 2;
  __bf16* wvt = (__bf16*)w; w += DD * 2;
  __bf16* wot = (__bf16*)w; w += DD * 2;
  __bf16* Qp  = (__bf16*)w; w += MSD * 2;
  __bf16* Kp  = (__bf16*)w; w += MSD * 2;
  __bf16* Vp  = (__bf16*)w; w += MSD * 2;
  __bf16* Ap  = xq;   // xq dead after the Q projection; reuse for attention output

  const int n4 = (int)(MSD / 4);
  dim3 cvtg((n4 + 255) / 256);
  cvt_f32_bf16<<<cvtg, 256, 0, stream>>>(q, xq, n4);
  cvt_f32_bf16<<<cvtg, 256, 0, stream>>>(k, xk, n4);
  cvt_f32_bf16<<<cvtg, 256, 0, stream>>>(v, xv, n4);

  dim3 tg(Dc / 16, Dc / 16), tb(16, 16);
  transpose_w<<<tg, tb, 0, stream>>>(Wq, wqt, Dc);
  transpose_w<<<tg, tb, 0, stream>>>(Wk, wkt, Dc);
  transpose_w<<<tg, tb, 0, stream>>>(Wv, wvt, Dc);
  transpose_w<<<tg, tb, 0, stream>>>(Wo, wot, Dc);

  dim3 gg(Dc / 128, (unsigned)(Mrows / 128));   // (8, 32)
  const float SCALE = 0.125f;                   // 1/sqrt(DK), folded into Q
  gemm_bf16<true><<<gg, 256, 0, stream>>>(xq, wqt, (void*)Qp, (int)Mrows, Dc, Dc, SCALE);
  gemm_bf16<true><<<gg, 256, 0, stream>>>(xk, wkt, (void*)Kp, (int)Mrows, Dc, Dc, 1.0f);
  gemm_bf16<true><<<gg, 256, 0, stream>>>(xv, wvt, (void*)Vp, (int)Mrows, Dc, Dc, 1.0f);

  flash_attn<<<dim3(Sc / 64, Bc * 16), 128, 0, stream>>>(Qp, Kp, Vp, Ap);

  gemm_bf16<false><<<gg, 256, 0, stream>>>(Ap, wot, d_out, (int)Mrows, Dc, Dc, 1.0f);
}
